// NeighborMessageAggregator_78065325572109
// MI455X (gfx1250) — matrix operand order repaired and checked by reading, not compiled, not run on hardware
//
#include <hip/hip_runtime.h>

typedef __attribute__((ext_vector_type(16))) __bf16 v16bf;
typedef __attribute__((ext_vector_type(8)))  float  v8f;
typedef __attribute__((ext_vector_type(4)))  unsigned int u32x4;

#define B_ROWS 1024
#define KNBR   16
#define HDIM   256
#define GDIM   2000
#define DDIM   4000
#define OUTW   4256
#define NFSTRIDE 4008   // bf16 elems; 8016B rows -> 16B aligned, 16-lane conflict-free
#define HSTRIDE  264    // bf16 elems; 528B rows  -> 16B aligned, conflict-free

union ABCast { u32x4 u[2]; v16bf v; };

__device__ __forceinline__ v16bf load_frag2(const __bf16* p0, const __bf16* p1) {
  ABCast c;
  c.u[0] = *(const u32x4*)p0;
  c.u[1] = *(const u32x4*)p1;
  return c.v;
}

// One M=16 row-tile GEMM: A (16 x 32*ntiles bf16, in LDS, row stride `astride`)
// times packed-B weights (32*ntiles x 256). Wave `wave` produces columns
// [wave*32, wave*32+32) as two 16x16 f32 accumulators.
__device__ __forceinline__ void gemm_rowtile(
    const __bf16* Alds, int astride, const __bf16* Wp, int ntiles,
    int lane, int wave, v8f& c0, v8f& c1)
{
  const int m   = lane & 15;
  const int hi8 = (lane >> 4) << 3;     // lanes 16..31 take K offset +8 (A layout)
  for (int t = 0; t < ntiles; ++t) {
    const __bf16* rp = Alds + m * astride + t * 32 + hi8;
    v16bf a = load_frag2(rp, rp + 16);  // K = {0..7} and {16..23} (+hi8)
    const __bf16* bb = Wp + (((size_t)(t * 16 + wave * 2) * 32 + lane) << 4);
    v16bf b0 = load_frag2(bb, bb + 8);
    v16bf b1 = load_frag2(bb + 512, bb + 520);
    c0 = __builtin_amdgcn_wmma_f32_16x16x32_bf16(false, a, false, b0, (short)0, c0, false, false);
    c1 = __builtin_amdgcn_wmma_f32_16x16x32_bf16(false, a, false, b1, (short)0, c1, false, false);
  }
}

// Pack a row-major f32 [D,256] weight into bf16 B-fragment order:
// idx = ((t*16 + g)*32 + L)*16 + e  <->  K = t*32 + (e&7) + 8*(L>=16) + 16*(e>=8), N = g*16 + (L&15)
__global__ __launch_bounds__(256) void pack_w_kernel(
    const float* __restrict__ W, __bf16* __restrict__ dst, int D)
{
  int i = blockIdx.x * 256 + threadIdx.x;
  if (i >= D * 256) return;
  int e = i & 15;
  int L = (i >> 4) & 31;
  int g = (i >> 9) & 15;
  int t = i >> 13;
  int kk = t * 32 + (e & 7) + ((L >> 4) << 3) + ((e >> 3) << 4);
  int h  = (g << 4) + (L & 15);
  dst[i] = (__bf16)W[(size_t)kk * 256 + h];
}

__global__ __launch_bounds__(256) void copy_enc_kernel(
    const float* __restrict__ enc, float* __restrict__ out)
{
  int i = blockIdx.x * 256 + threadIdx.x;
  if (i < B_ROWS * DDIM) {
    int b = i / DDIM, d = i % DDIM;
    out[(size_t)b * OUTW + d] = enc[i];
  }
}

// Per (anchor b, graph): gather 16 neighbor feature rows -> LDS bf16,
// WMMA attention hidden, tanh+Wa2 dot, softmax, weight-normalize,
// weighted feature sum -> xbar (f32, workspace).
__global__ __launch_bounds__(256) void attn_aggregate_kernel(
    const float* __restrict__ sp, const float* __restrict__ un,
    const int* __restrict__ idx1, const float* __restrict__ w1v,
    const int* __restrict__ idx2, const float* __restrict__ w2v,
    const __bf16* __restrict__ wsWa1,
    const float* __restrict__ ba1, const float* __restrict__ Wa2,
    const float* __restrict__ ba2,
    float* __restrict__ xbar1, float* __restrict__ xbar2)
{
  __shared__ __align__(16) __bf16 nf[16][NFSTRIDE];
  __shared__ float hid[16][257];
  __shared__ float satt[16], swn[16], swnn[16];

  const int b = blockIdx.x;
  const int graph = blockIdx.y;
  const int*   __restrict__ idx = graph ? idx2 : idx1;
  const float* __restrict__ wv  = graph ? w2v  : w1v;
  float*       __restrict__ xbar = graph ? xbar2 : xbar1;
  const int tid = threadIdx.x;

  for (int k = 0; k < KNBR; ++k) {
    const int row = idx[b * KNBR + k];
    const float* __restrict__ spr = sp + (size_t)row * GDIM;
    const float* __restrict__ unr = un + (size_t)row * GDIM;
    for (int d = tid; d < GDIM; d += 256) {
      nf[k][d]        = (__bf16)__logf(0.01f + spr[d]);
      nf[k][d + GDIM] = (__bf16)__logf(0.01f + unr[d]);
    }
  }
  __syncthreads();

  const int lane = tid & 31, wave = tid >> 5;
  v8f c0 = {}, c1 = {};
  gemm_rowtile(&nf[0][0], NFSTRIDE, wsWa1, DDIM / 32, lane, wave, c0, c1);

  const int mrow = (lane >> 4) << 3;
  const int h0 = wave * 32 + (lane & 15);
#pragma unroll
  for (int r = 0; r < 8; ++r) {
    const int M = r + mrow;
    hid[M][h0]      = tanhf(c0[r] + ba1[h0]);
    hid[M][h0 + 16] = tanhf(c1[r] + ba1[h0 + 16]);
  }
  __syncthreads();

  if (tid < 16) {
    float s = ba2[0];
    for (int h = 0; h < HDIM; ++h) s += hid[tid][h] * Wa2[h];
    satt[tid] = s;
  }
  __syncthreads();
  if (tid < 16) {
    float mx = satt[0];
    for (int i = 1; i < 16; ++i) mx = fmaxf(mx, satt[i]);
    float den = 0.f;
    for (int i = 0; i < 16; ++i) den += expf(satt[i] - mx);
    float att = expf(satt[tid] - mx) / den;
    swn[tid] = wv[b * KNBR + tid] * att;
  }
  __syncthreads();
  if (tid < 16) {
    float s = 0.f;
    for (int i = 0; i < 16; ++i) s += swn[i];
    swnn[tid] = swn[tid] / (s + 1e-12f);
  }
  __syncthreads();

  for (int d = tid; d < DDIM; d += 256) {
    float s = 0.f;
#pragma unroll
    for (int k = 0; k < KNBR; ++k) s += swnn[k] * (float)nf[k][d];
    xbar[(size_t)b * DDIM + d] = s;
  }
}

// Four batched [1024,4000]@[4000,256] GEMMs with ReLU+bias.
// mode 0: xbar1@W1 -> h1a   mode 1: xbar2@W1 -> h1b
// mode 2: (xbar1-enc)@Wd -> dout   mode 3: enc@Wr -> rout
__global__ __launch_bounds__(256) void big_gemm_kernel(
    const float* __restrict__ xbar1, const float* __restrict__ xbar2,
    const float* __restrict__ enc,
    const __bf16* __restrict__ wsW1, const __bf16* __restrict__ wsWd,
    const __bf16* __restrict__ wsWr,
    const float* __restrict__ b1, const float* __restrict__ bd,
    const float* __restrict__ br,
    float* __restrict__ h1a, float* __restrict__ h1b,
    float* __restrict__ dout, float* __restrict__ rout)
{
  __shared__ __align__(16) __bf16 Al[16][NFSTRIDE];
  const int mt = blockIdx.x, mode = blockIdx.y;
  const int tid = threadIdx.x;

  const float* __restrict__ src = (mode == 1) ? xbar2 : (mode == 3) ? enc : xbar1;
  for (int r = 0; r < 16; ++r) {
    const size_t base = (size_t)(mt * 16 + r) * DDIM;
    for (int d = tid; d < DDIM; d += 256) {
      float v = src[base + d];
      if (mode == 2) v -= enc[base + d];
      Al[r][d] = (__bf16)v;
    }
  }
  __syncthreads();

  const __bf16* W    = (mode <= 1) ? wsW1 : (mode == 2) ? wsWd : wsWr;
  const float*  bias = (mode <= 1) ? b1   : (mode == 2) ? bd   : br;
  float* outp = (mode == 0) ? h1a : (mode == 1) ? h1b : (mode == 2) ? dout : rout;

  const int lane = tid & 31, wave = tid >> 5;
  v8f c0 = {}, c1 = {};
  gemm_rowtile(&Al[0][0], NFSTRIDE, W, DDIM / 32, lane, wave, c0, c1);

  const int mrow = (lane >> 4) << 3;
  const int h0 = wave * 32 + (lane & 15);
#pragma unroll
  for (int r = 0; r < 8; ++r) {
    const int b = mt * 16 + r + mrow;
    outp[(size_t)b * HDIM + h0]      = fmaxf(c0[r] + bias[h0], 0.f);
    outp[(size_t)b * HDIM + h0 + 16] = fmaxf(c1[r] + bias[h0 + 16], 0.f);
  }
}

// H2a = relu(h1a@W2+b2), H2b = relu(h1b@W2+b2); cross-graph gate mix;
// + dout + rout; LayerNorm; sigmoid(message_gate) scale; write out[:,4000:].
__global__ __launch_bounds__(256) void final_combine_kernel(
    const float* __restrict__ h1a, const float* __restrict__ h1b,
    const __bf16* __restrict__ wsW2, const float* __restrict__ b2,
    const float* __restrict__ dout, const float* __restrict__ rout,
    const float* __restrict__ cg, const float* __restrict__ mg,
    const float* __restrict__ gamma, const float* __restrict__ beta,
    float* __restrict__ out)
{
  __shared__ __align__(16) __bf16 Al[16][HSTRIDE];
  __shared__ float proj[16][257];
  __shared__ float smu[16], srs[16];
  const int mt = blockIdx.x;
  const int tid = threadIdx.x, lane = tid & 31, wave = tid >> 5;

  for (int i = tid; i < 16 * HDIM; i += 256) {
    int r = i >> 8, h = i & 255;
    Al[r][h] = (__bf16)h1a[(size_t)(mt * 16 + r) * HDIM + h];
  }
  __syncthreads();
  v8f ca0 = {}, ca1 = {};
  gemm_rowtile(&Al[0][0], HSTRIDE, wsW2, HDIM / 32, lane, wave, ca0, ca1);
  __syncthreads();

  for (int i = tid; i < 16 * HDIM; i += 256) {
    int r = i >> 8, h = i & 255;
    Al[r][h] = (__bf16)h1b[(size_t)(mt * 16 + r) * HDIM + h];
  }
  __syncthreads();
  v8f cb0 = {}, cb1 = {};
  gemm_rowtile(&Al[0][0], HSTRIDE, wsW2, HDIM / 32, lane, wave, cb0, cb1);

  const int mrow = (lane >> 4) << 3;
  const int h0 = wave * 32 + (lane & 15);
#pragma unroll
  for (int r = 0; r < 8; ++r) {
    const int M = r + mrow;
    const int b = mt * 16 + M;
#pragma unroll
    for (int half = 0; half < 2; ++half) {
      const int h = h0 + 16 * half;
      float va = half ? ca1[r] : ca0[r];
      float vb = half ? cb1[r] : cb0[r];
      float h2a = fmaxf(va + b2[h], 0.f);
      float h2b = fmaxf(vb + b2[h], 0.f);
      float g = 1.f / (1.f + expf(-cg[h]));
      proj[M][h] = g * h2a + (1.f - g) * h2b
                 + dout[(size_t)b * HDIM + h] + rout[(size_t)b * HDIM + h];
    }
  }
  __syncthreads();

  if (tid < 16) {
    float s = 0.f, s2 = 0.f;
    for (int h = 0; h < HDIM; ++h) { float v = proj[tid][h]; s += v; s2 += v * v; }
    float mu = s * (1.f / 256.f);
    float var = s2 * (1.f / 256.f) - mu * mu;
    smu[tid] = mu;
    srs[tid] = rsqrtf(var + 1e-5f);
  }
  __syncthreads();

  for (int i = tid; i < 16 * HDIM; i += 256) {
    int r = i >> 8, h = i & 255;
    float v = (proj[r][h] - smu[r]) * srs[r] * gamma[h] + beta[h];
    float mgs = 1.f / (1.f + expf(-mg[h]));
    out[(size_t)(mt * 16 + r) * OUTW + DDIM + h] = mgs * v;
  }
}

extern "C" void kernel_launch(void* const* d_in, const int* in_sizes, int n_in,
                              void* d_out, int out_size, void* d_ws, size_t ws_size,
                              hipStream_t stream) {
  const float* enc  = (const float*)d_in[0];
  const int*   idx1 = (const int*)  d_in[1];
  const float* nw1  = (const float*)d_in[2];
  const int*   idx2 = (const int*)  d_in[3];
  const float* nw2  = (const float*)d_in[4];
  const float* sp   = (const float*)d_in[5];
  const float* un   = (const float*)d_in[6];
  const float* W1   = (const float*)d_in[7];
  const float* b1   = (const float*)d_in[8];
  const float* W2   = (const float*)d_in[9];
  const float* b2   = (const float*)d_in[10];
  const float* Wa1  = (const float*)d_in[11];
  const float* ba1  = (const float*)d_in[12];
  const float* Wa2  = (const float*)d_in[13];
  const float* ba2  = (const float*)d_in[14];
  const float* Wd   = (const float*)d_in[15];
  const float* bd   = (const float*)d_in[16];
  const float* Wr   = (const float*)d_in[17];
  const float* br   = (const float*)d_in[18];
  const float* gam  = (const float*)d_in[19];
  const float* bet  = (const float*)d_in[20];
  const float* mg   = (const float*)d_in[21];
  const float* cg   = (const float*)d_in[22];
  float* out = (float*)d_out;

  char* ws = (char*)d_ws;
  size_t o = 0;
  __bf16* wsWa1 = (__bf16*)(ws + o); o += (size_t)DDIM * 256 * 2;
  __bf16* wsW1  = (__bf16*)(ws + o); o += (size_t)DDIM * 256 * 2;
  __bf16* wsWd  = (__bf16*)(ws + o); o += (size_t)DDIM * 256 * 2;
  __bf16* wsWr  = (__bf16*)(ws + o); o += (size_t)DDIM * 256 * 2;
  __bf16* wsW2  = (__bf16*)(ws + o); o += (size_t)256 * 256 * 2;
  float* xbar1 = (float*)(ws + o); o += (size_t)B_ROWS * DDIM * 4;
  float* xbar2 = (float*)(ws + o); o += (size_t)B_ROWS * DDIM * 4;
  float* h1a   = (float*)(ws + o); o += (size_t)B_ROWS * HDIM * 4;
  float* h1b   = (float*)(ws + o); o += (size_t)B_ROWS * HDIM * 4;
  float* dout  = (float*)(ws + o); o += (size_t)B_ROWS * HDIM * 4;
  float* rout  = (float*)(ws + o); o += (size_t)B_ROWS * HDIM * 4;
  if (ws_size < o) return;  // workspace too small (needs ~45 MB)

  pack_w_kernel<<<(DDIM * 256) / 256, 256, 0, stream>>>(Wa1, wsWa1, DDIM);
  pack_w_kernel<<<(DDIM * 256) / 256, 256, 0, stream>>>(W1,  wsW1,  DDIM);
  pack_w_kernel<<<(DDIM * 256) / 256, 256, 0, stream>>>(Wd,  wsWd,  DDIM);
  pack_w_kernel<<<(DDIM * 256) / 256, 256, 0, stream>>>(Wr,  wsWr,  DDIM);
  pack_w_kernel<<<(256 * 256) / 256,  256, 0, stream>>>(W2,  wsW2,  256);

  copy_enc_kernel<<<(B_ROWS * DDIM + 255) / 256, 256, 0, stream>>>(enc, out);

  attn_aggregate_kernel<<<dim3(B_ROWS, 2), 256, 0, stream>>>(
      sp, un, idx1, nw1, idx2, nw2, wsWa1, ba1, Wa2, ba2, xbar1, xbar2);

  big_gemm_kernel<<<dim3(B_ROWS / 16, 4), 256, 0, stream>>>(
      xbar1, xbar2, enc, wsW1, wsWd, wsWr, b1, bd, br, h1a, h1b, dout, rout);

  final_combine_kernel<<<B_ROWS / 16, 256, 0, stream>>>(
      h1a, h1b, wsW2, b2, dout, rout, cg, mg, gam, bet, out);
}
